// SelfAttention_86483461472273
// MI455X (gfx1250) — matrix operand order repaired and checked by reading, not compiled, not run on hardware
//
#include <hip/hip_runtime.h>
#include <hip/hip_bf16.h>

typedef __bf16 v16bf __attribute__((ext_vector_type(16)));
typedef __bf16 v8bf  __attribute__((ext_vector_type(8)));
typedef __bf16 v4bf  __attribute__((ext_vector_type(4)));
typedef float  v8f   __attribute__((ext_vector_type(8)));
typedef int    v4i   __attribute__((ext_vector_type(4)));

union BF16Frag { v16bf v; v8bf h[2]; };

// ---- CDNA5 async global->LDS path (guarded; falls back to ds_store staging) ----
#if defined(__HIP_DEVICE_COMPILE__) && __has_builtin(__builtin_amdgcn_global_load_async_to_lds_b128)
#define HAVE_ASYNC_LDS 1
#else
#define HAVE_ASYNC_LDS 0
#endif

#if HAVE_ASYNC_LDS
typedef __attribute__((address_space(1))) v4i gv4i_t;   // global int4
typedef __attribute__((address_space(3))) v4i lv4i_t;   // LDS int4
__device__ __forceinline__ void async_copy16(const __bf16* g, __bf16* l) {
    // generic LDS pointer low 32 bits == LDS offset (ISA 10.2); global flat == global addr
    gv4i_t* gp = (gv4i_t*)(unsigned long long)(uintptr_t)g;
    lv4i_t* lp = (lv4i_t*)(unsigned int)(uintptr_t)l;
    __builtin_amdgcn_global_load_async_to_lds_b128(gp, lp, 0, 0);
}
#endif

__device__ __forceinline__ void wait_async0() {
#if HAVE_ASYNC_LDS
#if __has_builtin(__builtin_amdgcn_s_wait_asynccnt)
    __builtin_amdgcn_s_wait_asynccnt(0);
#else
    asm volatile("s_wait_asynccnt 0" ::: "memory");
#endif
#endif
}

// ---------------------------------------------------------------- converts
__global__ void cast_bf16_kernel(const float4* __restrict__ src, __bf16* __restrict__ dst, int n4) {
    int i = blockIdx.x * blockDim.x + threadIdx.x;
    if (i < n4) {
        float4 f = src[i];
        v4bf o; o.x = (__bf16)f.x; o.y = (__bf16)f.y; o.z = (__bf16)f.z; o.w = (__bf16)f.w;
        *(v4bf*)(dst + (size_t)i * 4) = o;
    }
}

// dst[c*R + r] = src[r*C + c]  (src: R x C f32 row-major -> dst: C x R bf16 row-major)
__global__ void transpose_bf16_kernel(const float* __restrict__ src, __bf16* __restrict__ dst, int R, int C) {
    int i = blockIdx.x * blockDim.x + threadIdx.x;
    if (i < R * C) {
        int r = i / C, c = i % C;
        dst[(size_t)c * R + r] = (__bf16)src[i];
    }
}

// ---------------------------------------------------------------- 128x128 bf16 GEMM core (K=2048)
// A: row-major (lda=2048). BT: transposed weights, row n = column n of W (ldb=2048).
// Double-buffered LDS staging, one barrier per 32-K step. Result -> LDS Ct[128][132] f32.
__device__ __forceinline__ void stage_tile(const __bf16* __restrict__ A, const __bf16* __restrict__ BT,
                                           int row0, int k0, __bf16* As, __bf16* Bs, int tid)
{
    const int r = tid >> 1;
    const int c = (tid & 1) << 4;
    const __bf16* ga = A  + (size_t)(row0 + r) * 2048 + k0 + c;
    const __bf16* gb = BT + (size_t)r * 2048 + k0 + c;
#if HAVE_ASYNC_LDS
    async_copy16(ga,     As + r * 40 + c);
    async_copy16(ga + 8, As + r * 40 + c + 8);
    async_copy16(gb,     Bs + r * 40 + c);
    async_copy16(gb + 8, Bs + r * 40 + c + 8);
#else
    *(uint4*)(As + r * 40 + c)     = *(const uint4*)ga;
    *(uint4*)(As + r * 40 + c + 8) = *(const uint4*)(ga + 8);
    *(uint4*)(Bs + r * 40 + c)     = *(const uint4*)gb;
    *(uint4*)(Bs + r * 40 + c + 8) = *(const uint4*)(gb + 8);
#endif
}

__device__ __forceinline__ void gemm128(const __bf16* __restrict__ A, const __bf16* __restrict__ BT,
                                        int row0, __bf16* As, __bf16* Bs, float* Ct)
{
    const int tid  = threadIdx.x;
    const int lane = tid & 31;
    const int w    = tid >> 5;
    const int wm   = w & 3;       // 4 waves along M (32 rows each)
    const int wn   = w >> 2;      // 2 waves along N (64 cols each)
    const int lr   = lane & 15;
    const int hi   = lane >> 4;   // 0: lanes 0-15, 1: lanes 16-31

    v8f acc[2][4] = {};

    stage_tile(A, BT, row0, 0, As, Bs, tid);     // prologue: tile 0 into buffer 0
    int cur = 0;

    for (int k0 = 0; k0 < 2048; k0 += 32) {
        wait_async0();
        __syncthreads();                          // buffer `cur` ready for all waves

        if (k0 + 32 < 2048)                       // prefetch next K-slice into other buffer
            stage_tile(A, BT, row0, k0 + 32, As + (cur ^ 1) * (128 * 40), Bs + (cur ^ 1) * (128 * 40), tid);

        const __bf16* Ac = As + cur * (128 * 40);
        const __bf16* Bc = Bs + cur * (128 * 40);

        BF16Frag af[2], bf[4];
#pragma unroll
        for (int i = 0; i < 2; i++) {             // A frag: lane=row, K chunks {0..7,16..23}/{8..15,24..31}
            const __bf16* p = Ac + (wm * 32 + i * 16 + lr) * 40;
            af[i].h[0] = *(const v8bf*)(p + hi * 8);
            af[i].h[1] = *(const v8bf*)(p + 16 + hi * 8);
        }
#pragma unroll
        for (int j = 0; j < 4; j++) {             // B frag: lane=col, K chunks {0..7,16..23}/{8..15,24..31}
            const __bf16* p = Bc + (wn * 64 + j * 16 + lr) * 40;
            bf[j].h[0] = *(const v8bf*)(p + hi * 8);
            bf[j].h[1] = *(const v8bf*)(p + 16 + hi * 8);
        }
#pragma unroll
        for (int i = 0; i < 2; i++)
#pragma unroll
            for (int j = 0; j < 4; j++)
                acc[i][j] = __builtin_amdgcn_wmma_f32_16x16x32_bf16(
                    false, af[i].v, false, bf[j].v, (short)0, acc[i][j], false, false);
        cur ^= 1;
    }
    __syncthreads();

    // dump accumulators to LDS (C layout: lane holds col n=lr, rows v + 8*hi)
#pragma unroll
    for (int i = 0; i < 2; i++)
#pragma unroll
        for (int j = 0; j < 4; j++) {
            int mb = wm * 32 + i * 16 + hi * 8;
            int n  = wn * 64 + j * 16 + lr;
#pragma unroll
            for (int v = 0; v < 8; v++) Ct[(mb + v) * 132 + n] = acc[i][j][v];
        }
    __syncthreads();
}

// ---------------------------------------------------------------- QKV projection + bias + RoPE
// grid: (32 row-tiles, 32 head-slots). gy<16 -> Q head, 16..23 -> K head, 24..31 -> V head.
__global__ __launch_bounds__(256) void qkv_gemm_kernel(
    const __bf16* __restrict__ xb, const __bf16* __restrict__ WqT,
    const __bf16* __restrict__ WkT, const __bf16* __restrict__ WvT,
    const float* __restrict__ bq, const float* __restrict__ bk, const float* __restrict__ bv,
    __bf16* __restrict__ qout, __bf16* __restrict__ kout, __bf16* __restrict__ vtout)
{
    __shared__ __align__(16) __bf16 As[2 * 128 * 40];
    __shared__ __align__(16) __bf16 Bs[2 * 128 * 40];
    __shared__ float Ct[128 * 132];

    int row0 = blockIdx.x * 128;
    int gy = blockIdx.y;
    int kind, head;
    const __bf16* BT;
    if (gy < 16)      { kind = 0; head = gy;      BT = WqT + (size_t)head * 128 * 2048; }
    else if (gy < 24) { kind = 1; head = gy - 16; BT = WkT + (size_t)head * 128 * 2048; }
    else              { kind = 2; head = gy - 24; BT = WvT + (size_t)head * 128 * 2048; }

    gemm128(xb, BT, row0, As, Bs, Ct);

    const float* bias = (kind == 0) ? bq : (kind == 1 ? bk : bv);
    for (int idx = threadIdx.x; idx < 128 * 128; idx += 256) {
        int m = idx >> 7, d = idx & 127;
        int grow = row0 + m;
        int b = grow >> 11, s = grow & 2047;
        float cval = Ct[m * 132 + d] + bias[head * 128 + d];
        if (kind == 2) {
            // V stored transposed: [b][hkv][d][s]
            vtout[((size_t)(b * 8 + head) * 128 + d) * 2048 + s] = (__bf16)cval;
        } else {
            int dj = d & 63;
            float invf = __expf(-(float)dj * (9.210340371976184f / 64.0f)); // theta^(-dj/64)
            float ang = (float)s * invf;
            float sn, cs; __sincosf(ang, &sn, &cs);
            int dother = (d < 64) ? d + 64 : d - 64;
            float oval = Ct[m * 132 + dother] + bias[head * 128 + dother];
            float y = (d < 64) ? (cval * cs - oval * sn) : (cval * cs + oval * sn);
            if (kind == 0) qout[((size_t)(b * 16 + head) * 2048 + s) * 128 + d] = (__bf16)y;
            else           kout[((size_t)(b * 8  + head) * 2048 + s) * 128 + d] = (__bf16)y;
        }
    }
}

// ---------------------------------------------------------------- flash attention (sliding window 1024)
// grid: B*H*(S/128) = 512 blocks, 8 waves/block, each wave owns a 16-row q tile.
__global__ __launch_bounds__(256) void attn_kernel(
    const __bf16* __restrict__ qb, const __bf16* __restrict__ kmat,
    const __bf16* __restrict__ vtb, __bf16* __restrict__ ctxb)
{
    __shared__ __align__(16) __bf16 Pl[8][16 * 40];

    const int tid  = threadIdx.x;
    const int lane = tid & 31;
    const int w    = tid >> 5;
    const int lr   = lane & 15;
    const int hi   = lane >> 4;

    int blk = blockIdx.x;
    int qt  = blk & 15;
    int bh  = blk >> 4;
    int h   = bh & 15;
    int b   = bh >> 4;
    int hkv = h >> 1;                // repeat(k,2) -> head h uses kv head h/2
    int q0  = qt * 128 + w * 16;

    const __bf16* qbase = qb   + ((size_t)(b * 16 + h)   * 2048 + q0) * 128;
    const __bf16* kbase = kmat + ((size_t)(b * 8 + hkv)  * 2048) * 128;
    const __bf16* vbase = vtb  + ((size_t)(b * 8 + hkv)  * 128) * 2048;
    __bf16* Prow = &Pl[w][0];

    // preload 4 Q fragments (K-dim chunks of 32 along D=128)
    BF16Frag qf[4];
#pragma unroll
    for (int kk = 0; kk < 4; kk++) {
        const __bf16* p = qbase + (size_t)lr * 128 + kk * 32;
        qf[kk].h[0] = *(const v8bf*)(p + hi * 8);
        qf[kk].h[1] = *(const v8bf*)(p + 16 + hi * 8);
    }

    v8f acc[8] = {};
    float rmax[8], rsum[8];
#pragma unroll
    for (int v = 0; v < 8; v++) { rmax[v] = -1e30f; rsum[v] = 0.0f; }

    const float scale = 0.08838834764831845f;   // 128^-0.5
    int klo = q0 - 1023; if (klo < 0) klo = 0; klo &= ~31;
    int khi = q0 + 15;

    for (int kb = klo; kb <= khi; kb += 32) {
        // scores: 16 q rows x 32 keys = two 16x16 C tiles, 4 WMMA each over D
        v8f sc[2] = {};
#pragma unroll
        for (int nb = 0; nb < 2; nb++) {
#pragma unroll
            for (int kk = 0; kk < 4; kk++) {
                BF16Frag kf;
                const __bf16* p = kbase + (size_t)(kb + nb * 16 + lr) * 128 + kk * 32;
                kf.h[0] = *(const v8bf*)(p + hi * 8);
                kf.h[1] = *(const v8bf*)(p + 16 + hi * 8);
                sc[nb] = __builtin_amdgcn_wmma_f32_16x16x32_bf16(
                    false, qf[kk].v, false, kf.v, (short)0, sc[nb], false, false);
            }
        }

        // online softmax (rows spread over 16-lane groups; VGPR v -> row v + 8*hi)
        float fac[8];
#pragma unroll
        for (int v = 0; v < 8; v++) {
            int m = v + hi * 8;
            int qpos = q0 + m;
            float s0 = sc[0][v] * scale, s1 = sc[1][v] * scale;
            int key0 = kb + lr, key1 = kb + 16 + lr;
            bool v0 = (key0 <= qpos) && (key0 > qpos - 1024);
            bool v1 = (key1 <= qpos) && (key1 > qpos - 1024);
            float mt = fmaxf(v0 ? s0 : -1e30f, v1 ? s1 : -1e30f);
#pragma unroll
            for (int off = 1; off < 16; off <<= 1) mt = fmaxf(mt, __shfl_xor(mt, off, 16));
            float mn = fmaxf(rmax[v], mt);
            float f  = __expf(rmax[v] - mn);
            rmax[v]  = mn;
            float p0 = v0 ? __expf(s0 - mn) : 0.0f;
            float p1 = v1 ? __expf(s1 - mn) : 0.0f;
            float ps = p0 + p1;
#pragma unroll
            for (int off = 1; off < 16; off <<= 1) ps += __shfl_xor(ps, off, 16);
            rsum[v] = rsum[v] * f + ps;
            fac[v]  = f;
            Prow[m * 40 + lr]      = (__bf16)p0;   // C-layout -> row-major P in LDS
            Prow[m * 40 + 16 + lr] = (__bf16)p1;
        }

        // rescale running context
#pragma unroll
        for (int j = 0; j < 8; j++)
#pragma unroll
            for (int v = 0; v < 8; v++) acc[j][v] *= fac[v];

        asm volatile("s_wait_dscnt 0" ::: "memory");   // LDS RAW: P writes -> A-frag reads (same wave)

        // P as A fragment (16x32)
        BF16Frag pf;
        {
            const __bf16* p = Prow + lr * 40;
            pf.h[0] = *(const v8bf*)(p + hi * 8);
            pf.h[1] = *(const v8bf*)(p + 16 + hi * 8);
        }
        // ctx += P * V, V read from transposed layout (contiguous along keys)
#pragma unroll
        for (int j = 0; j < 8; j++) {
            BF16Frag vf;
            const __bf16* p = vbase + (size_t)(j * 16 + lr) * 2048 + kb;
            vf.h[0] = *(const v8bf*)(p + hi * 8);
            vf.h[1] = *(const v8bf*)(p + 16 + hi * 8);
            acc[j] = __builtin_amdgcn_wmma_f32_16x16x32_bf16(
                false, pf.v, false, vf.v, (short)0, acc[j], false, false);
        }
    }

    // normalize and write ctx as bf16 rows for the output GEMM: [b][s][h*128+d]
#pragma unroll
    for (int v = 0; v < 8; v++) {
        int m = v + hi * 8;
        int s = q0 + m;
        float inv = 1.0f / rsum[v];
#pragma unroll
        for (int j = 0; j < 8; j++) {
            int d = j * 16 + lr;
            ctxb[(size_t)(b * 2048 + s) * 2048 + h * 128 + d] = (__bf16)(acc[j][v] * inv);
        }
    }
}

// ---------------------------------------------------------------- output projection
__global__ __launch_bounds__(256) void out_gemm_kernel(
    const __bf16* __restrict__ ctxb, const __bf16* __restrict__ WoT,
    const float* __restrict__ bo, float* __restrict__ out)
{
    __shared__ __align__(16) __bf16 As[2 * 128 * 40];
    __shared__ __align__(16) __bf16 Bs[2 * 128 * 40];
    __shared__ float Ct[128 * 132];

    int row0 = blockIdx.x * 128;
    int col0 = blockIdx.y * 128;
    gemm128(ctxb, WoT + (size_t)col0 * 2048, row0, As, Bs, Ct);

    for (int idx = threadIdx.x; idx < 128 * 128; idx += 256) {
        int m = idx >> 7, d = idx & 127;
        out[(size_t)(row0 + m) * 2048 + col0 + d] = Ct[m * 132 + d] + bo[col0 + d];
    }
}

// ---------------------------------------------------------------- launch
extern "C" void kernel_launch(void* const* d_in, const int* in_sizes, int n_in,
                              void* d_out, int out_size, void* d_ws, size_t ws_size,
                              hipStream_t stream)
{
    const float* x  = (const float*)d_in[0];
    const float* Wq = (const float*)d_in[1];
    const float* bq = (const float*)d_in[2];
    const float* Wk = (const float*)d_in[3];
    const float* bk = (const float*)d_in[4];
    const float* Wv = (const float*)d_in[5];
    const float* bv = (const float*)d_in[6];
    const float* Wo = (const float*)d_in[7];
    const float* bo = (const float*)d_in[8];
    float* out = (float*)d_out;

    __bf16* p   = (__bf16*)d_ws;
    __bf16* xb  = p; p += 8388608;   // (B*S, F) bf16
    __bf16* WqT = p; p += 4194304;   // (2048, F)
    __bf16* WkT = p; p += 2097152;   // (1024, F)
    __bf16* WvT = p; p += 2097152;   // (1024, F)
    __bf16* WoT = p; p += 4194304;   // (F, 2048)
    __bf16* qb  = p; p += 8388608;   // (B, H, S, D) post-RoPE
    __bf16* kb  = p; p += 4194304;   // (B, HKV, S, D) post-RoPE
    __bf16* vtb = p; p += 4194304;   // (B, HKV, D, S) transposed
    __bf16* ctx = p; p += 8388608;   // (B*S, H*D)

    cast_bf16_kernel<<<8388608 / 1024, 256, 0, stream>>>((const float4*)x, xb, 8388608 / 4);
    transpose_bf16_kernel<<<(2048 * 2048) / 256, 256, 0, stream>>>(Wq, WqT, 2048, 2048);
    transpose_bf16_kernel<<<(2048 * 1024) / 256, 256, 0, stream>>>(Wk, WkT, 2048, 1024);
    transpose_bf16_kernel<<<(2048 * 1024) / 256, 256, 0, stream>>>(Wv, WvT, 2048, 1024);
    transpose_bf16_kernel<<<(2048 * 2048) / 256, 256, 0, stream>>>(Wo, WoT, 2048, 2048);

    qkv_gemm_kernel<<<dim3(32, 32), 256, 0, stream>>>(xb, WqT, WkT, WvT, bq, bk, bv, qb, kb, vtb);
    attn_kernel<<<512, 256, 0, stream>>>(qb, kb, vtb, ctx);
    out_gemm_kernel<<<dim3(32, 16), 256, 0, stream>>>(ctx, WoT, bo, out);
}
